// LinkPredictor_42863773614195
// MI455X (gfx1250) — matrix operand order repaired and checked by reading, not compiled, not run on hardware
//
#include <hip/hip_runtime.h>
#include <hip/hip_bf16.h>

// ---------------------------------------------------------------------------
// Edge-MLP link predictor for MI455X (gfx1250, wave32, WMMA).
//
// score[e] = relu(concat(feat[src[e]],feat[dst[e]]) @ W1^T + b1) @ W2^T + b2
//
// Layer 1 (105 GFLOP) runs on v_wmma_f32_16x16x32_bf16 (bf16 in, f32 acc).
// Layer 2 (0.8 GFLOP) runs on VALU + wave shuffles.
// W1^T staged once per workgroup into 64 KB of LDS as bf16, stored
// FRAGMENT-MAJOR: each wave's B-fragment is two contiguous ds_load_b128s.
// Node table (51 MB) is L2-resident (192 MB L2), so the 1.6 GB of random
// feature gathers stream from L2, not HBM.
// ---------------------------------------------------------------------------

typedef __attribute__((ext_vector_type(16))) __bf16 v16bf;
typedef __attribute__((ext_vector_type(8)))  float  v8f;

#define DIM              128
#define TWO_D            256
#define WAVES_PER_BLOCK  8
#define TILES_PER_WAVE   4
#define TILES_PER_BLOCK  (WAVES_PER_BLOCK * TILES_PER_WAVE)
#define BLOCK_THREADS    (WAVES_PER_BLOCK * 32)

union Frag {
    v16bf v;
    unsigned u[8];
    struct { uint4 lo; uint4 hi; } q;
};

// round-half-up f32->bf16 pair, packed with one v_perm_b32 (3 VALU total)
__device__ __forceinline__ unsigned pack_bf16x2(float lo, float hi) {
    unsigned ahi = __builtin_bit_cast(unsigned, hi) + 0x8000u;
    unsigned alo = __builtin_bit_cast(unsigned, lo) + 0x8000u;
    // out = {ahi[31:24], ahi[23:16], alo[31:24], alo[23:16]}
    return __builtin_amdgcn_perm(ahi, alo, 0x07060302u);
}

__global__ __launch_bounds__(BLOCK_THREADS, 3)   // <=341 VGPRs: no acc spills
void edge_mlp_wmma_kernel(const float* __restrict__ node_feat,
                          const int*   __restrict__ src,
                          const int*   __restrict__ dst,
                          const float* __restrict__ w1,   // [128][256] row-major (out,in)
                          const float* __restrict__ b1,   // [128]
                          const float* __restrict__ w2,   // [2][128]
                          const float* __restrict__ b2,   // [2]
                          float*       __restrict__ out,  // [E][2]
                          int n_edges, int n_tiles)
{
    // W1^T as bf16, FRAGMENT-MAJOR (bijective permutation of the 16384 dwords):
    //   dword (n, d)  [n = out channel, d = k/2]
    //   ct = n>>4, l15 = n&15, kk = d>>4, hi16 = (d>>3)&1, v = d&7
    //   lane = hi16*16 + l15, vb = v>>2, j = v&3
    //   idx = ((kk*8+ct)*2 + vb)*128 + lane*4 + j
    // => per (kk,ct) a lane reads dwords [idx..idx+3] and [idx+128..idx+131]:
    //    two ds_load_b128, lane stride 4 dwords -> lanes 0-15 hit all 64 banks,
    //    lanes 16-31 repeat them: the 2-cycle minimum for 128 dwords.
    __shared__ unsigned wlds[128 * 128];   // 64 KB exactly

    const int tid = threadIdx.x;

    // ---- stage + convert W1 -> LDS (b64 float2 reads, perm-packed bf16) ----
    const float2* w1v = reinterpret_cast<const float2*>(w1);
    for (int i = tid; i < 128 * 128; i += BLOCK_THREADS) {
        int n = i >> 7;        // output channel
        int d = i & 127;       // dword index = k/2
        float2 f = w1v[i];     // == (w1[n*256 + 2d], w1[n*256 + 2d + 1])
        int ct  = n >> 4, l15n = n & 15;
        int kk  = d >> 4, r    = d & 15;
        int hi  = r >> 3, v    = r & 7;
        int ln  = hi * 16 + l15n;
        int idx = ((((kk << 3) + ct) << 1) + (v >> 2)) * 128 + (ln << 2) + (v & 3);
        wlds[idx] = pack_bf16x2(f.x, f.y);
    }
    __syncthreads();

    const int lane = tid & 31;
    const int wave = tid >> 5;
    const int l15  = lane & 15;
    const int hi16 = lane >> 4;     // which 16-lane half of the wave

    // single LDS base pointer; all 128 fragment loads use immediate offsets
    const unsigned* fbase = wlds + (lane << 2);

    // per-lane loop-invariant weights (column N = ct*16 + l15)
    float b1v[8], w2v0[8], w2v1[8];
    #pragma unroll
    for (int ct = 0; ct < 8; ++ct) {
        b1v[ct]  = b1[ct * 16 + l15];
        w2v0[ct] = w2[ct * 16 + l15];
        w2v1[ct] = w2[DIM + ct * 16 + l15];
    }
    const float b2_0 = b2[0], b2_1 = b2[1];

    for (int t = 0; t < TILES_PER_WAVE; ++t) {
        const int tile = blockIdx.x * TILES_PER_BLOCK + wave * TILES_PER_WAVE + t;
        if (tile >= n_tiles) break;            // wave-uniform: EXEC stays full
        const long long tile_base = (long long)tile * 16;

        // A-matrix row owned by this lane (ISA 16-bit A layout: lanes 0-15 and
        // 16-31 both hold rows 0-15, split across K). Clamp => EXEC all-ones.
        long long e = tile_base + l15;
        if (e >= n_edges) e = n_edges - 1;
        const float* srow = node_feat + (long long)src[e] * DIM;
        const float* drow = node_feat + (long long)dst[e] * DIM;

        v8f acc[8];
        #pragma unroll
        for (int ct = 0; ct < 8; ++ct)
            #pragma unroll
            for (int r = 0; r < 8; ++r)
                acc[ct][r] = b1v[ct];          // bias folded into accumulator

        #pragma unroll
        for (int kk = 0; kk < 8; ++kk) {       // K step of 32 (k = kk*32..+31)
            // kk 0..3 -> src half of concat, kk 4..7 -> dst half
            const float* rowp = (kk < 4 ? srow : drow) + (kk & 3) * 32 + hi16 * 16;

            // A fragment: u[v] holds K pair; v<8 -> rowp[2v..2v+1] (rowp[0..15])
            const float4* rp = reinterpret_cast<const float4*>(rowp);
            float4 x0 = rp[0], x1 = rp[1], x2 = rp[2], x3 = rp[3];  // 4x b128
            Frag a;
            a.u[0] = pack_bf16x2(x0.x, x0.y);
            a.u[1] = pack_bf16x2(x0.z, x0.w);
            a.u[2] = pack_bf16x2(x1.x, x1.y);
            a.u[3] = pack_bf16x2(x1.z, x1.w);
            a.u[4] = pack_bf16x2(x2.x, x2.y);
            a.u[5] = pack_bf16x2(x2.z, x2.w);
            a.u[6] = pack_bf16x2(x3.x, x3.y);
            a.u[7] = pack_bf16x2(x3.z, x3.w);

            #pragma unroll
            for (int ct = 0; ct < 8; ++ct) {
                const int fo = (((kk << 3) + ct) << 8);   // fragment dword offset
                Frag bf;
                bf.q.lo = *reinterpret_cast<const uint4*>(fbase + fo);
                bf.q.hi = *reinterpret_cast<const uint4*>(fbase + fo + 128);
                acc[ct] = __builtin_amdgcn_wmma_f32_16x16x32_bf16(
                    false, a.v, false, bf.v, (short)0, acc[ct], false, false);
            }
        }

        // ---- layer 2: relu + [128]->2, per-lane partials over this lane's N ----
        float p0[8], p1[8];
        #pragma unroll
        for (int r = 0; r < 8; ++r) { p0[r] = 0.f; p1[r] = 0.f; }
        #pragma unroll
        for (int ct = 0; ct < 8; ++ct) {
            const float wa = w2v0[ct], wb = w2v1[ct];
            #pragma unroll
            for (int r = 0; r < 8; ++r) {
                float h = fmaxf(acc[ct][r], 0.0f);   // edge M = r + 8*hi16
                p0[r] = fmaf(h, wa, p0[r]);
                p1[r] = fmaf(h, wb, p1[r]);
            }
        }
        // reduce over the 16 lanes of each half (N mod 16)
        #pragma unroll
        for (int mask = 1; mask <= 8; mask <<= 1) {
            #pragma unroll
            for (int r = 0; r < 8; ++r) {
                p0[r] += __shfl_xor(p0[r], mask, 32);
                p1[r] += __shfl_xor(p1[r], mask, 32);
            }
        }
        // lane l15 (<8) of each half writes edge tile_base + hi16*8 + l15
        float o0 = p0[0], o1 = p1[0];
        #pragma unroll
        for (int r = 1; r < 8; ++r) {
            if (l15 == r) { o0 = p0[r]; o1 = p1[r]; }   // v_cndmask select
        }
        if (l15 < 8) {
            long long eo = tile_base + hi16 * 8 + l15;
            if (eo < n_edges) {
                float2 val = make_float2(o0 + b2_0, o1 + b2_1);
                *reinterpret_cast<float2*>(out + eo * 2) = val;
            }
        }
    }
}

extern "C" void kernel_launch(void* const* d_in, const int* in_sizes, int n_in,
                              void* d_out, int out_size, void* d_ws, size_t ws_size,
                              hipStream_t stream) {
    const float* node_feat = (const float*)d_in[0];
    const int*   src       = (const int*)d_in[1];
    const int*   dst       = (const int*)d_in[2];
    const float* w1        = (const float*)d_in[3];
    const float* b1        = (const float*)d_in[4];
    const float* w2        = (const float*)d_in[5];
    const float* b2        = (const float*)d_in[6];
    float*       out       = (float*)d_out;

    const int n_edges = in_sizes[1];
    if (n_edges <= 0) return;
    const int n_tiles = (n_edges + 15) / 16;
    const int blocks  = (n_tiles + TILES_PER_BLOCK - 1) / TILES_PER_BLOCK;

    hipLaunchKernelGGL(edge_mlp_wmma_kernel, dim3(blocks), dim3(BLOCK_THREADS), 0, stream,
                       node_feat, src, dst, w1, b1, w2, b2, out, n_edges, n_tiles);
}
